// Attention_74096775791016
// MI455X (gfx1250) — matrix operand order repaired and checked by reading, not compiled, not run on hardware
//
#include <hip/hip_runtime.h>
#include <cstdint>

#define B_ 4
#define H_ 16
#define S_ 2048
#define D_ 64

typedef __attribute__((ext_vector_type(16))) _Float16 v16h;
typedef __attribute__((ext_vector_type(8)))  float    v8f;
typedef __attribute__((ext_vector_type(4)))  unsigned int u32x4;
typedef __attribute__((ext_vector_type(4)))  int  i32x4;
typedef __attribute__((ext_vector_type(8)))  int  i32x8;

// ---------------------------------------------------------------------------
// Kernel 1: raw scores  ATT[b,h,q,k] = (1/8) * sum_d Q[b,h,q,d] * K[b,h,k,d]
// One wave computes a 16x128 strip: A (16x32 f16) reused across 8 B tiles.
// WMMA layouts (wave32, CDNA5 ISA):
//   A 16x32 f16 : lane = 16*half + m ; element e -> K = e + (e>=8 ? 8 : 0) + half*8
//   B 32x16 f16 : lane = 16*half + n ; element e -> K = e + (e>=8 ? 8 : 0) + half*8
//   C/D 16x16 f32: lane = 16*half + n ; VGPR r -> M = r + 8*half
// ---------------------------------------------------------------------------
__global__ void qk_kernel(const float* __restrict__ Q,
                          const float* __restrict__ Kmat,
                          float* __restrict__ ATT) {
  const int lane  = threadIdx.x;       // 0..31
  const int n     = lane & 15;
  const int half  = lane >> 4;
  const int qBase = (blockIdx.x * blockDim.y + threadIdx.y) * 16;
  const int kBase = blockIdx.y * 128;
  const int bh    = blockIdx.z;

  const float* __restrict__ Qb = Q    + (size_t)bh * S_ * D_;
  const float* __restrict__ Kb = Kmat + (size_t)bh * S_ * D_;
  float*       __restrict__ Ab = ATT  + (size_t)bh * S_ * S_;

  v8f acc[8];
#pragma unroll
  for (int t = 0; t < 8; ++t) acc[t] = (v8f){0.f,0.f,0.f,0.f,0.f,0.f,0.f,0.f};

#pragma unroll
  for (int kc = 0; kc < D_; kc += 32) {
    v16h a;
    const float* qrow = Qb + (size_t)(qBase + n) * D_ + kc + half * 8;
#pragma unroll
    for (int e = 0; e < 8; ++e)  a[e] = (_Float16)qrow[e];
#pragma unroll
    for (int e = 8; e < 16; ++e) a[e] = (_Float16)qrow[e + 8];

#pragma unroll
    for (int t = 0; t < 8; ++t) {
      v16h b;
      const float* krow = Kb + (size_t)(kBase + t * 16 + n) * D_ + kc + half * 8;
#pragma unroll
      for (int e = 0; e < 16; ++e) b[e] = (_Float16)krow[e < 8 ? e : e + 8];
      acc[t] = __builtin_amdgcn_wmma_f32_16x16x32_f16(
          false, a, false, b, (short)0, acc[t], false, false);
    }
  }

  const float scale = 0.125f;  // 1/sqrt(64)
#pragma unroll
  for (int t = 0; t < 8; ++t) {
#pragma unroll
    for (int r = 0; r < 8; ++r) {
      Ab[(size_t)(qBase + r + 8 * half) * S_ + (kBase + t * 16 + n)] = acc[t][r] * scale;
    }
  }
}

// ---------------------------------------------------------------------------
// Kernel 2: softmax over the HEAD axis, vectorized float4 (16B coalesced b128).
// One thread handles 4 consecutive k for one (b,q); 16 head values per lane
// live in registers (stride between heads = S*S elements).
// ---------------------------------------------------------------------------
__global__ void softmax_h_kernel(float* __restrict__ ATT) {
  const size_t SS   = (size_t)S_ * S_;
  const size_t idx  = ((size_t)blockIdx.x * blockDim.x + threadIdx.x) * 4;  // over B*S*S
  const size_t b    = idx / SS;
  const size_t qk   = idx - b * SS;
  const size_t base = b * (size_t)H_ * SS + qk;

  float v[H_][4];
#pragma unroll
  for (int h = 0; h < H_; ++h) {
    const float4 t = *(const float4*)(ATT + base + (size_t)h * SS);
    v[h][0] = t.x; v[h][1] = t.y; v[h][2] = t.z; v[h][3] = t.w;
  }
#pragma unroll
  for (int c = 0; c < 4; ++c) {
    float m = -3.0e38f;
#pragma unroll
    for (int h = 0; h < H_; ++h) m = fmaxf(m, v[h][c]);
    float s = 0.f;
#pragma unroll
    for (int h = 0; h < H_; ++h) { v[h][c] = __expf(v[h][c] - m); s += v[h][c]; }
    const float inv = 1.0f / s;
#pragma unroll
    for (int h = 0; h < H_; ++h) v[h][c] *= inv;
  }
#pragma unroll
  for (int h = 0; h < H_; ++h) {
    float4 t; t.x = v[h][0]; t.y = v[h][1]; t.z = v[h][2]; t.w = v[h][3];
    *(float4*)(ATT + base + (size_t)h * SS) = t;
  }
}

// ---------------------------------------------------------------------------
// TDM: DMA one 32x64 f32 tile of V (8 KB) into LDS (CDNA5 ISA).
// D# group0: count=1 | lds_addr | global_addr[31:0] | global_addr[56:32]+type=2
// D# group1: data_size=4B, tensor_dim0=64, tensor_dim1=2048, tile=64x32,
//            dim0_stride=64, dim1_stride=2048*64 (=0x20000)
// Tracked by TENSORcnt; waited with s_wait_tensorcnt.
// ---------------------------------------------------------------------------
__device__ __forceinline__ void tdm_load_v_tile(const float* gsrc, void* lds_dst) {
  const unsigned long long ga = (unsigned long long)(uintptr_t)gsrc;
  const unsigned lds_addr = (unsigned)(uintptr_t)lds_dst;  // LDS aperture: low 32 bits
  u32x4 g0 = { 1u,                                     // count=1 (valid user D#)
               lds_addr,                               // lds_addr
               (unsigned)(ga & 0xFFFFFFFFu),           // global_addr[31:0]
               (unsigned)((ga >> 32) & 0x01FFFFFFu) | 0x80000000u };  // [56:32] | type=2
  i32x8 g1 = { (int)0x00020000,        // workgroup_mask=0, data_size=2 (4 bytes)
               (int)(64u   << 16),     // tensor_dim0[15:0]  = 64   (bits 79:64 zero)
               (int)(2048u << 16),     // tensor_dim1[15:0]  = 2048
               (int)(64u   << 16),     // tile_dim0 = 64
               (int)32,                // tile_dim1 = 32
               (int)64,                // tensor_dim0_stride[31:0] = 64
               0,                      // dim0_stride[47:32]=0, dim1_stride[15:0]=0
               2 };                    // tensor_dim1_stride[47:16] = 2  (0x20000 total)
  i32x4 gz4 = {0, 0, 0, 0};
#if defined(__clang_major__) && (__clang_major__ >= 23)
  i32x8 gz8 = {0, 0, 0, 0, 0, 0, 0, 0};
  __builtin_amdgcn_tensor_load_to_lds(g0, g1, gz4, gz4, gz8, 0);
#else
  __builtin_amdgcn_tensor_load_to_lds(g0, g1, gz4, gz4, 0);
#endif
}

// ---------------------------------------------------------------------------
// Kernel 3: Z = ATT @ V per (b,h): [S,S] x [S,D].
// Block = 16 waves covering a 256-row Q strip. V is staged into LDS by the TDM
// in 32x64 tiles, double-buffered: wave 0 issues chunk i+1, waits TENSORcnt<=1
// (chunk i complete, TDM ops are in-order per wave), barrier, all waves compute
// chunk i from LDS. ATT rows stream from global (read exactly once).
// ---------------------------------------------------------------------------
__global__ void av_kernel(const float* __restrict__ ATT,
                          const float* __restrict__ V,
                          float* __restrict__ Z) {
  __shared__ float vbuf[2][32 * D_];   // 2 x 8 KB double buffer

  const int lane  = threadIdx.x;
  const int n     = lane & 15;
  const int half  = lane >> 4;
  const int wave  = threadIdx.y;       // 0..15
  const int qBase = (blockIdx.x * 16 + wave) * 16;
  const int bh    = blockIdx.y;

  const float* __restrict__ Ab = ATT + (size_t)bh * S_ * S_;
  const float* __restrict__ Vb = V   + (size_t)bh * S_ * D_;
  float*       __restrict__ Zb = Z   + (size_t)bh * S_ * D_;

  v8f acc[4];
#pragma unroll
  for (int t = 0; t < 4; ++t) acc[t] = (v8f){0.f,0.f,0.f,0.f,0.f,0.f,0.f,0.f};

  if (wave == 0) tdm_load_v_tile(Vb, &vbuf[0][0]);

  const int NCHUNK = S_ / 32;          // 64
  for (int i = 0; i < NCHUNK; ++i) {
    const int kc = i * 32;
    if (wave == 0) {
      if (i < NCHUNK - 1) {
        // prefetch chunk i+1 into the other buffer (its readers finished at
        // the barrier that ended iteration i-1), then wait for chunk i.
        tdm_load_v_tile(Vb + (size_t)(kc + 32) * D_, &vbuf[(i + 1) & 1][0]);
        __builtin_amdgcn_s_wait_tensorcnt(1);
      } else {
        __builtin_amdgcn_s_wait_tensorcnt(0);
      }
    }
    __syncthreads();

    const float* vl = vbuf[i & 1];

    // A fragment: 16 attention values per lane from global (row qBase+n)
    v16h a;
    const float* arow = Ab + (size_t)(qBase + n) * S_ + kc + half * 8;
#pragma unroll
    for (int e = 0; e < 8; ++e)  a[e] = (_Float16)arow[e];
#pragma unroll
    for (int e = 8; e < 16; ++e) a[e] = (_Float16)arow[e + 8];

#pragma unroll
    for (int t = 0; t < 4; ++t) {
      v16h b;
#pragma unroll
      for (int e = 0; e < 16; ++e)
        b[e] = (_Float16)vl[(half * 8 + e + (e >= 8 ? 8 : 0)) * D_ + t * 16 + n];
      acc[t] = __builtin_amdgcn_wmma_f32_16x16x32_f16(
          false, a, false, b, (short)0, acc[t], false, false);
    }
    __syncthreads();
  }

#pragma unroll
  for (int t = 0; t < 4; ++t) {
#pragma unroll
    for (int r = 0; r < 8; ++r) {
      Zb[(size_t)(qBase + r + 8 * half) * D_ + (t * 16 + n)] = acc[t][r];
    }
  }
}

// ---------------------------------------------------------------------------
extern "C" void kernel_launch(void* const* d_in, const int* in_sizes, int n_in,
                              void* d_out, int out_size, void* d_ws, size_t ws_size,
                              hipStream_t stream) {
  (void)in_sizes; (void)n_in; (void)out_size; (void)d_ws; (void)ws_size;

  const float* Q  = (const float*)d_in[0];
  const float* K  = (const float*)d_in[1];
  const float* Vv = (const float*)d_in[2];

  float* Z   = (float*)d_out;                       // [B,H,S,D]
  float* ATT = Z + (size_t)B_ * H_ * S_ * D_;       // [B,H,S,S]

  // 1) raw scaled scores into ATT
  qk_kernel<<<dim3(S_ / 64, S_ / 128, B_ * H_), dim3(32, 4), 0, stream>>>(Q, K, ATT);

  // 2) softmax across heads, in place (float4 per thread)
  const int smblocks = (B_ * S_ * S_ / 4) / 256;    // 16384
  softmax_h_kernel<<<smblocks, 256, 0, stream>>>(ATT);

  // 3) Z = ATT @ V with TDM-staged V tiles
  av_kernel<<<dim3(S_ / 256, B_ * H_), dim3(32, 16), 0, stream>>>(ATT, Vv, Z);
}
